// GraphAttentionUpdate_87196426043573
// MI455X (gfx1250) — matrix operand order repaired and checked by reading, not compiled, not run on hardware
//
#include <hip/hip_runtime.h>

// ---------------------------------------------------------------------------
// GraphAttentionUpdate for MI455X (gfx1250, wave32, WMMA).
// All four 64x64 projections run through v_wmma_f32_16x16x32_f16 tiles with
// fp32 accumulation; scatter-softmax / scatter-sum use fp32 L2 atomics.
// ---------------------------------------------------------------------------

typedef __attribute__((ext_vector_type(16))) _Float16 v16h;
typedef __attribute__((ext_vector_type(8)))  float    v8f;

constexpr int   Ln         = 100000;    // nodes (divisible by 16)
constexpr int   En         = 1200000;   // edges (divisible by 16)
constexpr int   Dm         = 64;        // d_model
constexpr int   Hh         = 4;         // heads (tile == head, DH==16)
constexpr float SCALE      = 0.25f;     // DH^-0.5, DH=16
constexpr float EPS_SM     = 1e-7f;
constexpr float EPS_LN     = 1e-5f;

// ---- workspace layout (in floats) ----
constexpr size_t WS_Q    = 0;                       // [L,64]
constexpr size_t WS_LOG  = WS_Q   + (size_t)Ln*Dm;  // [E,4]  logits -> exp
constexpr size_t WS_MENC = WS_LOG + (size_t)En*Hh;  // [L,4]  encoded max (u32)
constexpr size_t WS_S    = WS_MENC+ (size_t)Ln*Hh;  // [L,4]  softmax denom
constexpr size_t WS_AGG  = WS_S   + (size_t)Ln*Hh;  // [L,64] aggregated values

// Monotonic float<->uint encoding so atomicMax(u32) implements float max.
__device__ inline unsigned enc_f(float f) {
  unsigned u = __float_as_uint(f);
  return (u & 0x80000000u) ? ~u : (u | 0x80000000u);
}
__device__ inline float dec_f(unsigned u) {
  return __uint_as_float((u & 0x80000000u) ? (u ^ 0x80000000u) : ~u);
}

// ---- WMMA fragment loaders (contiguous k order shared by A and B, wave32) ----
// A 16x32 f16: lane holds row M=lane&15; elements 0..7 -> K = kbase+16*sel+j,
// elements 8..15 -> K = kbase+16*sel+8+j, sel = lane>>4.
__device__ inline v16h load_a_frag_f32(const float* __restrict__ src,
                                       int row0, int kbase, int lane) {
  const int row = row0 + (lane & 15);
  const int sel = (lane >> 4) & 1;
  const float* p0 = src + (size_t)row * Dm + kbase + sel * 16;
  const float* p1 = p0 + 8;
  v16h a;
#pragma unroll
  for (int j = 0; j < 8; ++j) a[j]     = (_Float16)p0[j];
#pragma unroll
  for (int j = 0; j < 8; ++j) a[8 + j] = (_Float16)p1[j];
  return a;
}

// B 32x16 f16: lane holds col N=lane&15; elements j -> K = kbase+16*sel+j.
// W is [64(K),64(N)] row-major.
__device__ inline v16h load_b_frag_f32(const float* __restrict__ W,
                                       int tile, int kbase, int lane) {
  const int col  = tile * 16 + (lane & 15);
  const int krow = kbase + ((lane >> 4) & 1) * 16;
  v16h b;
#pragma unroll
  for (int j = 0; j < 16; ++j) b[j] = (_Float16)W[(size_t)(krow + j) * Dm + col];
  return b;
}

__device__ inline v8f wmma_f16(v16h a, v16h b, v8f c) {
  return __builtin_amdgcn_wmma_f32_16x16x32_f16(
      /*neg_a=*/false, a, /*neg_b=*/false, b,
      /*c_mod=*/(short)0, c, /*reuse_a=*/false, /*reuse_b=*/false);
}

// ---------------------------------------------------------------------------
// Kernel 0: init segment-max (encoded -inf == 0), denom, agg accumulators.
// ---------------------------------------------------------------------------
__global__ void k_init(unsigned* __restrict__ menc, float* __restrict__ s,
                       float* __restrict__ agg) {
  const int i = blockIdx.x * blockDim.x + threadIdx.x;
  if (i < Ln * Hh) { menc[i] = 0u; s[i] = 0.f; }
  if (i < Ln * Dm) agg[i] = 0.f;
}

// ---------------------------------------------------------------------------
// Kernel 1: Y = X @ W + b   (rows x 64 @ 64x64), one 16-row tile per wave.
// ---------------------------------------------------------------------------
__global__ void __launch_bounds__(256)
k_gemm_bias(const float* __restrict__ X, const float* __restrict__ W,
            const float* __restrict__ bias, float* __restrict__ Y, int nrows) {
  const int lane   = threadIdx.x & 31;
  const int wave   = (blockIdx.x * blockDim.x + threadIdx.x) >> 5;
  const int nwaves = (gridDim.x * blockDim.x) >> 5;
  const int sel    = (lane >> 4) & 1;
  const int n      = lane & 15;

  v16h bf[2][4];
#pragma unroll
  for (int ks = 0; ks < 2; ++ks)
#pragma unroll
    for (int t = 0; t < 4; ++t) bf[ks][t] = load_b_frag_f32(W, t, ks * 32, lane);
  float bv[4];
#pragma unroll
  for (int t = 0; t < 4; ++t) bv[t] = bias[t * 16 + n];

  const int ntiles = nrows >> 4;
  for (int tile = wave; tile < ntiles; tile += nwaves) {
    const int row0 = tile * 16;
    v16h a0 = load_a_frag_f32(X, row0, 0, lane);
    v16h a1 = load_a_frag_f32(X, row0, 32, lane);
#pragma unroll
    for (int t = 0; t < 4; ++t) {
      v8f c = {};
      c = wmma_f16(a0, bf[0][t], c);
      c = wmma_f16(a1, bf[1][t], c);
#pragma unroll
      for (int r = 0; r < 8; ++r) {
        const int m = r + sel * 8;
        Y[(size_t)(row0 + m) * Dm + t * 16 + n] = c[r] + bv[t];
      }
    }
  }
}

// ---------------------------------------------------------------------------
// Kernel 2: fused k-projection + per-head dot with gathered q + segment max.
// logit[e,h] = SCALE * dot(q[idx[e], h*16:..], (k_in @ Wk + bk)[e, h*16:..])
// ---------------------------------------------------------------------------
__global__ void __launch_bounds__(256)
k_logits(const float* __restrict__ Kin, const float* __restrict__ Wk,
         const float* __restrict__ bk, const float* __restrict__ q,
         const int* __restrict__ idx, float* __restrict__ logit,
         unsigned* __restrict__ menc) {
  const int lane   = threadIdx.x & 31;
  const int wave   = (blockIdx.x * blockDim.x + threadIdx.x) >> 5;
  const int nwaves = (gridDim.x * blockDim.x) >> 5;
  const int sel    = (lane >> 4) & 1;
  const int n      = lane & 15;

  v16h bf[2][4];
#pragma unroll
  for (int ks = 0; ks < 2; ++ks)
#pragma unroll
    for (int t = 0; t < 4; ++t) bf[ks][t] = load_b_frag_f32(Wk, t, ks * 32, lane);
  float bv[4];
#pragma unroll
  for (int t = 0; t < 4; ++t) bv[t] = bk[t * 16 + n];

  const int ntiles = En >> 4;
  for (int tile = wave; tile < ntiles; tile += nwaves) {
    const int e0 = tile * 16;
    v16h a0 = load_a_frag_f32(Kin, e0, 0, lane);
    v16h a1 = load_a_frag_f32(Kin, e0, 32, lane);
    int nid[8];
#pragma unroll
    for (int r = 0; r < 8; ++r) nid[r] = idx[e0 + r + sel * 8];

#pragma unroll
    for (int t = 0; t < 4; ++t) {  // t == head
      v8f c = {};
      c = wmma_f16(a0, bf[0][t], c);
      c = wmma_f16(a1, bf[1][t], c);
#pragma unroll
      for (int r = 0; r < 8; ++r) {
        float p = (c[r] + bv[t]) * q[(size_t)nid[r] * Dm + t * 16 + n];
        // reduce over the 16 head columns (lanes within each half-wave)
#pragma unroll
        for (int o = 1; o < 16; o <<= 1) p += __shfl_xor(p, o, 16);
        if (n == 0) {
          const int m   = r + sel * 8;
          const float d = SCALE * p;
          logit[(size_t)(e0 + m) * Hh + t] = d;
          atomicMax(&menc[(size_t)nid[r] * Hh + t], enc_f(d));
        }
      }
    }
  }
}

// ---------------------------------------------------------------------------
// Kernel 3: ex = exp(logit - m[idx]) in place; s[idx] += ex.
// ---------------------------------------------------------------------------
__global__ void k_exp(const int* __restrict__ idx, float* __restrict__ ex,
                      const unsigned* __restrict__ menc, float* __restrict__ s) {
  const int i = blockIdx.x * blockDim.x + threadIdx.x;
  if (i >= En * Hh) return;
  const int e = i >> 2, h = i & 3;
  const int nid = idx[e];
  const float m = dec_f(menc[(size_t)nid * Hh + h]);
  const float v = __expf(ex[i] - m);
  ex[i] = v;
  atomicAdd(&s[(size_t)nid * Hh + h], v);
}

// ---------------------------------------------------------------------------
// Kernel 4: fused v-projection * softmax weight + scatter-sum into agg.
// ---------------------------------------------------------------------------
__global__ void __launch_bounds__(256)
k_vagg(const float* __restrict__ Vin, const float* __restrict__ Wv,
       const float* __restrict__ bvb, const int* __restrict__ idx,
       const float* __restrict__ ex, const float* __restrict__ s,
       float* __restrict__ agg) {
  const int lane   = threadIdx.x & 31;
  const int wave   = (blockIdx.x * blockDim.x + threadIdx.x) >> 5;
  const int nwaves = (gridDim.x * blockDim.x) >> 5;
  const int sel    = (lane >> 4) & 1;
  const int n      = lane & 15;

  v16h bf[2][4];
#pragma unroll
  for (int ks = 0; ks < 2; ++ks)
#pragma unroll
    for (int t = 0; t < 4; ++t) bf[ks][t] = load_b_frag_f32(Wv, t, ks * 32, lane);
  float bv[4];
#pragma unroll
  for (int t = 0; t < 4; ++t) bv[t] = bvb[t * 16 + n];

  const int ntiles = En >> 4;
  for (int tile = wave; tile < ntiles; tile += nwaves) {
    const int e0 = tile * 16;
    v16h a0 = load_a_frag_f32(Vin, e0, 0, lane);
    v16h a1 = load_a_frag_f32(Vin, e0, 32, lane);
    int nid[8];
#pragma unroll
    for (int r = 0; r < 8; ++r) nid[r] = idx[e0 + r + sel * 8];

#pragma unroll
    for (int t = 0; t < 4; ++t) {  // t == head
      v8f c = {};
      c = wmma_f16(a0, bf[0][t], c);
      c = wmma_f16(a1, bf[1][t], c);
#pragma unroll
      for (int r = 0; r < 8; ++r) {
        const int m  = r + sel * 8;
        const float w = ex[(size_t)(e0 + m) * Hh + t] /
                        (s[(size_t)nid[r] * Hh + t] + EPS_SM);
        atomicAdd(&agg[(size_t)nid[r] * Dm + t * 16 + n], (c[r] + bv[t]) * w);
      }
    }
  }
}

// ---------------------------------------------------------------------------
// Kernel 5: out = LayerNorm(agg @ Wo + bo) * gamma + beta, LN fused in-register
// via width-16 butterfly reductions over the WMMA C-fragment rows.
// ---------------------------------------------------------------------------
__global__ void __launch_bounds__(256)
k_out(const float* __restrict__ agg, const float* __restrict__ Wo,
      const float* __restrict__ bo, const float* __restrict__ gamma,
      const float* __restrict__ beta, float* __restrict__ out) {
  const int lane   = threadIdx.x & 31;
  const int wave   = (blockIdx.x * blockDim.x + threadIdx.x) >> 5;
  const int nwaves = (gridDim.x * blockDim.x) >> 5;
  const int sel    = (lane >> 4) & 1;
  const int n      = lane & 15;

  v16h bf[2][4];
#pragma unroll
  for (int ks = 0; ks < 2; ++ks)
#pragma unroll
    for (int t = 0; t < 4; ++t) bf[ks][t] = load_b_frag_f32(Wo, t, ks * 32, lane);
  float bv[4], gm[4], bt[4];
#pragma unroll
  for (int t = 0; t < 4; ++t) {
    bv[t] = bo[t * 16 + n];
    gm[t] = gamma[t * 16 + n];
    bt[t] = beta[t * 16 + n];
  }

  const int ntiles = Ln >> 4;
  for (int tile = wave; tile < ntiles; tile += nwaves) {
    const int row0 = tile * 16;
    v16h a0 = load_a_frag_f32(agg, row0, 0, lane);
    v16h a1 = load_a_frag_f32(agg, row0, 32, lane);
    v8f c[4];
#pragma unroll
    for (int t = 0; t < 4; ++t) {
      c[t] = v8f{};
      c[t] = wmma_f16(a0, bf[0][t], c[t]);
      c[t] = wmma_f16(a1, bf[1][t], c[t]);
    }
#pragma unroll
    for (int r = 0; r < 8; ++r) {
      float px = 0.f, psq = 0.f;
#pragma unroll
      for (int t = 0; t < 4; ++t) {
        const float x = c[t][r] + bv[t];
        px += x; psq += x * x;
      }
#pragma unroll
      for (int o = 1; o < 16; o <<= 1) {
        px  += __shfl_xor(px,  o, 16);
        psq += __shfl_xor(psq, o, 16);
      }
      const float mu   = px * (1.f / 64.f);
      const float var  = psq * (1.f / 64.f) - mu * mu;
      const float rstd = rsqrtf(var + EPS_LN);
      const int m = r + sel * 8;
#pragma unroll
      for (int t = 0; t < 4; ++t) {
        const float x = c[t][r] + bv[t];
        out[(size_t)(row0 + m) * Dm + t * 16 + n] = gm[t] * (x - mu) * rstd + bt[t];
      }
    }
  }
}

// ---------------------------------------------------------------------------
extern "C" void kernel_launch(void* const* d_in, const int* in_sizes, int n_in,
                              void* d_out, int out_size, void* d_ws, size_t ws_size,
                              hipStream_t stream) {
  (void)in_sizes; (void)n_in; (void)out_size; (void)ws_size;
  const float* f    = (const float*)d_in[0];
  const float* k_in = (const float*)d_in[1];
  const float* v_in = (const float*)d_in[2];
  const int*   idx  = (const int*)d_in[3];
  const float* Wq   = (const float*)d_in[4];
  const float* bq   = (const float*)d_in[5];
  const float* Wk   = (const float*)d_in[6];
  const float* bk   = (const float*)d_in[7];
  const float* Wv   = (const float*)d_in[8];
  const float* bvb  = (const float*)d_in[9];
  const float* Wo   = (const float*)d_in[10];
  const float* bo   = (const float*)d_in[11];
  const float* gmm  = (const float*)d_in[12];
  const float* bta  = (const float*)d_in[13];
  float* out = (float*)d_out;

  float*    ws    = (float*)d_ws;
  float*    q     = ws + WS_Q;
  float*    logit = ws + WS_LOG;
  unsigned* menc  = (unsigned*)(ws + WS_MENC);
  float*    s     = ws + WS_S;
  float*    agg   = ws + WS_AGG;

  k_init<<<(Ln * Dm + 255) / 256, 256, 0, stream>>>(menc, s, agg);
  k_gemm_bias<<<512, 256, 0, stream>>>(f, Wq, bq, q, Ln);
  k_logits<<<1024, 256, 0, stream>>>(k_in, Wk, bk, q, idx, logit, menc);
  k_exp<<<(En * Hh + 255) / 256, 256, 0, stream>>>(idx, logit, menc, s);
  k_vagg<<<1024, 256, 0, stream>>>(v_in, Wv, bvb, idx, logit, s, agg);
  k_out<<<512, 256, 0, stream>>>(agg, Wo, bo, gmm, bta, out);
}
